// GraphCNNSAT_38792144617931
// MI455X (gfx1250) — matrix operand order repaired and checked by reading, not compiled, not run on hardware
//
#include <hip/hip_runtime.h>
#include <hip/hip_bf16.h>

#define HH 128

typedef __attribute__((ext_vector_type(16))) _Float16 v16h;
typedef __attribute__((ext_vector_type(4)))  _Float16 v4h;
typedef __attribute__((ext_vector_type(8)))  float    v8f;

// Native CDNA5 f32 global atomic add (avoid any chance of a CAS-loop lowering).
__device__ __forceinline__ void atomic_add_f32(float* p, float v) {
    asm volatile("global_atomic_add_f32 %0, %1, off" :: "v"(p), "v"(v) : "memory");
}

// ---------------- layer 0 (D_IN = 1) ----------------

__global__ void k_init_pooled0(const float* __restrict__ vf, const float* __restrict__ eps,
                               float* __restrict__ p0, int n) {
    int i = blockIdx.x * blockDim.x + threadIdx.x;
    if (i < n) p0[i] = (1.f + eps[0]) * vf[i];
}

__global__ void k_scatter0(const float* __restrict__ vf, const int* __restrict__ src,
                           const int* __restrict__ dst, float* __restrict__ p0, int E) {
    int e = blockIdx.x * blockDim.x + threadIdx.x;
    if (e < E) atomic_add_f32(&p0[dst[e]], vf[src[e]]);
}

__global__ void k_reduce0(const float* __restrict__ p, float* __restrict__ out2, int n) {
    float s = 0.f, q = 0.f;
    for (int i = blockIdx.x * blockDim.x + threadIdx.x; i < n; i += gridDim.x * blockDim.x) {
        float v = p[i]; s += v; q += v * v;
    }
    for (int off = 16; off; off >>= 1) { s += __shfl_xor(s, off, 32); q += __shfl_xor(q, off, 32); }
    __shared__ float ls[8], lq[8];
    int wave = threadIdx.x >> 5, lane = threadIdx.x & 31;
    if (lane == 0) { ls[wave] = s; lq[wave] = q; }
    __syncthreads();
    if (threadIdx.x == 0) {
        float ts = 0.f, tq = 0.f;
        for (int w = 0; w < 8; ++w) { ts += ls[w]; tq += lq[w]; }
        atomic_add_f32(&out2[0], ts);
        atomic_add_f32(&out2[1], tq);
    }
}

// fold Linear(1->128) + BN(g1,bb1) into x = relu(p*a[j] + c[j])
__global__ void k_finalize0(const float* __restrict__ W1, const float* __restrict__ b1,
                            const float* __restrict__ g1, const float* __restrict__ bb1,
                            const float* __restrict__ ss, float* __restrict__ aOut,
                            float* __restrict__ cOut, float invN) {
    int j = threadIdx.x;
    float m = ss[0] * invN;
    float var = ss[1] * invN - m * m;
    float w = W1[j];
    float meanj = m * w + b1[j];
    float varj = var * w * w;
    float s = g1[j] * rsqrtf(varj + 1e-5f);
    aOut[j] = w * s;
    cOut[j] = (b1[j] - meanj) * s + bb1[j];
}

__global__ void k_build_x0(const float* __restrict__ p0, const float* __restrict__ aC,
                           const float* __restrict__ cC, _Float16* __restrict__ x0, int total) {
    int idx = blockIdx.x * blockDim.x + threadIdx.x;
    int base = idx * 4;
    if (base >= total) return;
    int nrow = base >> 7;
    int j0 = base & 127;
    float p = p0[nrow];
    v4h o;
    #pragma unroll
    for (int t = 0; t < 4; ++t) {
        float v = fmaf(p, aC[j0 + t], cC[j0 + t]);
        v = v > 0.f ? v : 0.f;
        o[t] = (_Float16)v;
    }
    *(v4h*)(x0 + base) = o;
}

// ---------------- fused GEMM (+optional input BN/ReLU) + column stats ----------------
// C[128-row tile x 128] = act(A * S + T) @ W + bias ; accumulates colsum/colsumsq.
// Uses v_wmma_f32_16x16x32_f16; A/B fragments packed in LDS in contiguous k order (k = 16*(lane/16) + idx).

__global__ __launch_bounds__(256) void k_gemm_bn(
    const _Float16* __restrict__ Ain, const float* __restrict__ W,
    const float* __restrict__ bias, const float* __restrict__ colS,
    const float* __restrict__ colT, int applyAct,
    _Float16* __restrict__ Cout, float* __restrict__ colSum, float* __restrict__ colSsq) {
    __shared__ _Float16 aPack[16384];  // 8 row-tiles x 4 k-blocks, 32 lanes x 16 halves each
    __shared__ _Float16 wPack[16384];  // 4 k-blocks  x 8 n-tiles
    int tid = threadIdx.x;
    int row0 = blockIdx.x * 128;

    // pack W (f32 global -> f16 LDS, B-fragment layout: lane = (k/16)*16 + n%16, idx = k%16)
    for (int e = tid; e < HH * HH; e += 256) {
        int k = e >> 7, n = e & 127;
        int kb = k >> 5, kk = k & 31, nt = n >> 4, nn = n & 15;
        int lane = ((kk >> 4) << 4) | nn;
        int idx = kk & 15;
        wPack[(((kb * 8 + nt) * 32 + lane) << 4) + idx] = (_Float16)W[e];
    }
    // pack A with optional per-column scale/shift + relu
    for (int it = 0; it < 16; ++it) {
        int q = (it * 256 + tid) * 4;
        int r = q >> 7, k0 = q & 127;
        v4h h4 = *(const v4h*)(Ain + (size_t)(row0 + r) * HH + k0);
        if (applyAct) {
            #pragma unroll
            for (int t = 0; t < 4; ++t) {
                float f = fmaf((float)h4[t], colS[k0 + t], colT[k0 + t]);
                f = f > 0.f ? f : 0.f;
                h4[t] = (_Float16)f;
            }
        }
        int rt = r >> 4, m = r & 15;
        #pragma unroll
        for (int t = 0; t < 4; ++t) {
            int k = k0 + t;
            int kb = k >> 5, kk = k & 31;
            int hi = kk >> 4;
            int idx = kk & 15;
            int lane = (hi << 4) | m;
            aPack[(((rt * 4 + kb) * 32 + lane) << 4) + idx] = h4[t];
        }
    }
    __syncthreads();

    int wave = tid >> 5, lane = tid & 31;
    v8f acc[8] = {};
    #pragma unroll
    for (int kb = 0; kb < 4; ++kb) {
        v16h af = *(const v16h*)(aPack + (((wave * 4 + kb) * 32 + lane) << 4));
        #pragma unroll
        for (int nt = 0; nt < 8; ++nt) {
            v16h bf = *(const v16h*)(wPack + (((kb * 8 + nt) * 32 + lane) << 4));
            acc[nt] = __builtin_amdgcn_wmma_f32_16x16x32_f16(
                false, af, false, bf, (short)0, acc[nt], false, false);
        }
    }

    __syncthreads();  // reuse aPack LDS for block-level column stats
    float* bS = (float*)aPack;
    float* bQ = bS + HH;
    if (tid < HH) { bS[tid] = 0.f; bQ[tid] = 0.f; }
    __syncthreads();

    int nn = lane & 15, hi = lane >> 4;
    #pragma unroll
    for (int nt = 0; nt < 8; ++nt) {
        int col = nt * 16 + nn;
        float b = bias[col];
        float psum = 0.f, pssq = 0.f;
        #pragma unroll
        for (int r = 0; r < 8; ++r) {
            float v = acc[nt][r] + b;
            int row = row0 + wave * 16 + hi * 8 + r;
            Cout[(size_t)row * HH + col] = (_Float16)v;
            psum += v;
            pssq += v * v;
        }
        atomicAdd(&bS[col], psum);   // LDS: ds_add_f32
        atomicAdd(&bQ[col], pssq);
    }
    __syncthreads();
    if (tid < HH) {
        atomic_add_f32(&colSum[tid], bS[tid]);
        atomic_add_f32(&colSsq[tid], bQ[tid]);
    }
}

// stats -> per-column scale/shift; re-zero accumulators for the next GEMM
__global__ void k_finalize_bn(const float* __restrict__ g, const float* __restrict__ b,
                              float* __restrict__ colSum, float* __restrict__ colSsq,
                              float* __restrict__ S, float* __restrict__ T, float invN) {
    int j = threadIdx.x;
    float m = colSum[j] * invN;
    float var = colSsq[j] * invN - m * m;
    float s = g[j] * rsqrtf(var + 1e-5f);
    S[j] = s;
    T[j] = b[j] - m * s;
    colSum[j] = 0.f;
    colSsq[j] = 0.f;
}

// h = relu(y*S + T); also pre-init pooled_next = (1+eps)*h for the next layer's scatter
__global__ void k_update_h(const _Float16* __restrict__ y2, const float* __restrict__ S,
                           const float* __restrict__ T, _Float16* __restrict__ hOut,
                           _Float16* __restrict__ pool, const float* __restrict__ eps,
                           int epsIdx, int writePool, int total) {
    int idx = blockIdx.x * blockDim.x + threadIdx.x;
    int base = idx * 4;
    if (base >= total) return;
    int j0 = base & 127;
    v4h y = *(const v4h*)(y2 + base);
    float f = 0.f;
    if (writePool) f = 1.f + eps[epsIdx];
    v4h hv, pv;
    #pragma unroll
    for (int t = 0; t < 4; ++t) {
        float v = fmaf((float)y[t], S[j0 + t], T[j0 + t]);
        v = v > 0.f ? v : 0.f;
        hv[t] = (_Float16)v;
        pv[t] = (_Float16)(f * v);
    }
    *(v4h*)(hOut + base) = hv;
    if (writePool) *(v4h*)(pool + base) = pv;
}

// ---------------- message passing: one wave per edge, pk f16 atomics ----------------

__global__ __launch_bounds__(256) void k_scatter_f16(const _Float16* __restrict__ h,
                                                     const int* __restrict__ src,
                                                     const int* __restrict__ dst,
                                                     _Float16* __restrict__ pooled, int E) {
    int gid = blockIdx.x * blockDim.x + threadIdx.x;
    int e = gid >> 5;
    int lane = gid & 31;
    if (e >= E) return;
    e = __builtin_amdgcn_readfirstlane(e);  // wave-uniform edge -> scalar index loads
    int s = src[e], d = dst[e];
    uint2 v = *(const uint2*)(h + (size_t)s * HH + lane * 4);
    _Float16* p0 = pooled + (size_t)d * HH + lane * 4;
    asm volatile("global_atomic_pk_add_f16 %0, %1, off" :: "v"(p0), "v"(v.x) : "memory");
    asm volatile("global_atomic_pk_add_f16 %0, %1, off offset:4" :: "v"(p0), "v"(v.y) : "memory");
}

// ---------------- classifier head: logits(128->2) + softmax, one wave per node ------

__global__ __launch_bounds__(256) void k_head(const _Float16* __restrict__ h,
                                              const float* __restrict__ fcW,
                                              const float* __restrict__ fcb,
                                              float* __restrict__ out, int n) {
    int gid = blockIdx.x * blockDim.x + threadIdx.x;
    int node = gid >> 5;
    int lane = gid & 31;
    if (node >= n) return;
    v4h hv = *(const v4h*)(h + (size_t)node * HH + lane * 4);
    float d0 = 0.f, d1 = 0.f;
    #pragma unroll
    for (int t = 0; t < 4; ++t) {
        float hf = (float)hv[t];
        int k = lane * 4 + t;
        d0 = fmaf(hf, fcW[k * 2 + 0], d0);
        d1 = fmaf(hf, fcW[k * 2 + 1], d1);
    }
    for (int off = 16; off; off >>= 1) { d0 += __shfl_xor(d0, off, 32); d1 += __shfl_xor(d1, off, 32); }
    if (lane == 0) {
        d0 += fcb[0];
        d1 += fcb[1];
        float mx = fmaxf(d0, d1);
        float e0 = expf(d0 - mx), e1 = expf(d1 - mx);
        float inv = 1.f / (e0 + e1);
        out[(size_t)node * 2 + 0] = e0 * inv;
        out[(size_t)node * 2 + 1] = e1 * inv;
    }
}

// ---------------- host orchestration ----------------

extern "C" void kernel_launch(void* const* d_in, const int* in_sizes, int n_in,
                              void* d_out, int out_size, void* d_ws, size_t ws_size,
                              hipStream_t stream) {
    (void)n_in; (void)out_size; (void)ws_size;
    const float* var_feat = (const float*)d_in[1];
    const int*   esrc     = (const int*)d_in[2];
    const int*   edst     = (const int*)d_in[3];
    const float* eps      = (const float*)d_in[4];
    const float* m0W1     = (const float*)d_in[5];
    const float* m0b1     = (const float*)d_in[6];
    const float* m0g1     = (const float*)d_in[7];
    const float* m0bb1    = (const float*)d_in[8];
    const float* m0W2     = (const float*)d_in[9];
    const float* m0b2     = (const float*)d_in[10];
    const float* bn0g     = (const float*)d_in[11];
    const float* bn0b     = (const float*)d_in[12];
    const float* mW1      = (const float*)d_in[13];
    const float* mb1      = (const float*)d_in[14];
    const float* mg1      = (const float*)d_in[15];
    const float* mbb1     = (const float*)d_in[16];
    const float* mW2      = (const float*)d_in[17];
    const float* mb2      = (const float*)d_in[18];
    const float* bng      = (const float*)d_in[19];
    const float* bnb      = (const float*)d_in[20];
    const float* fcW      = (const float*)d_in[21];
    const float* fcb      = (const float*)d_in[22];
    float* out = (float*)d_out;

    const int N = in_sizes[1];      // 400000 (D_IN = 1)
    const int E = in_sizes[2];      // 6400000
    const int total = N * HH;       // 51.2M f16 elements per buffer
    const float invN = 1.f / (float)N;

    char* ws = (char*)d_ws;
    const size_t bufBytes = (size_t)total * sizeof(_Float16);   // 102,400,000
    _Float16* bufA  = (_Float16*)ws;                      // pooled (f16 accum) / y2
    _Float16* bufB  = (_Float16*)(ws + bufBytes);         // x / y1
    _Float16* bufC  = (_Float16*)(ws + 2 * bufBytes);     // h
    float* pooled0  = (float*)(ws + 3 * bufBytes);        // N f32
    float* stats    = (float*)(ws + 3 * bufBytes + ((size_t)N * 4 + 255) / 256 * 256);
    float* colSum = stats;          // 128
    float* colSsq = stats + 128;    // 128
    float* Svec   = stats + 256;    // 128
    float* Tvec   = stats + 384;    // 128
    float* a0     = stats + 512;    // 128
    float* c0     = stats + 640;    // 128
    float* ss0    = stats + 768;    // 2

    hipMemsetAsync(stats, 0, 1024 * sizeof(float), stream);

    const int gElem4 = (total / 4 + 255) / 256;     // elementwise, 4 elems/thread
    const int gGemm  = N / 128;                     // 3125
    const int gEdge  = (E * 32 + 255) / 256;        // 1 wave per edge
    const int gNode  = (N * 32 + 255) / 256;        // 1 wave per node

    // ---- layer 0 ----
    k_init_pooled0<<<(N + 255) / 256, 256, 0, stream>>>(var_feat, eps, pooled0, N);
    k_scatter0<<<(E + 255) / 256, 256, 0, stream>>>(var_feat, esrc, edst, pooled0, E);
    k_reduce0<<<1024, 256, 0, stream>>>(pooled0, ss0, N);
    k_finalize0<<<1, 128, 0, stream>>>(m0W1, m0b1, m0g1, m0bb1, ss0, a0, c0, invN);
    k_build_x0<<<gElem4, 256, 0, stream>>>(pooled0, a0, c0, bufB, total);
    k_gemm_bn<<<gGemm, 256, 0, stream>>>(bufB, m0W2, m0b2, nullptr, nullptr, 0,
                                         bufA, colSum, colSsq);
    k_finalize_bn<<<1, 128, 0, stream>>>(bn0g, bn0b, colSum, colSsq, Svec, Tvec, invN);
    k_update_h<<<gElem4, 256, 0, stream>>>(bufA, Svec, Tvec, bufC, bufA, eps, 1, 1, total);

    // ---- layers 1..3 ----
    for (int l = 1; l <= 3; ++l) {
        int w = l - 1;
        k_scatter_f16<<<gEdge, 256, 0, stream>>>(bufC, esrc, edst, bufA, E);
        k_gemm_bn<<<gGemm, 256, 0, stream>>>(bufA, mW1 + (size_t)w * HH * HH, mb1 + w * HH,
                                             nullptr, nullptr, 0, bufB, colSum, colSsq);
        k_finalize_bn<<<1, 128, 0, stream>>>(mg1 + w * HH, mbb1 + w * HH,
                                             colSum, colSsq, Svec, Tvec, invN);
        k_gemm_bn<<<gGemm, 256, 0, stream>>>(bufB, mW2 + (size_t)w * HH * HH, mb2 + w * HH,
                                             Svec, Tvec, 1, bufA, colSum, colSsq);
        k_finalize_bn<<<1, 128, 0, stream>>>(bng + w * HH, bnb + w * HH,
                                             colSum, colSsq, Svec, Tvec, invN);
        k_update_h<<<gElem4, 256, 0, stream>>>(bufA, Svec, Tvec, bufC, bufA, eps,
                                               l + 1, (l < 3) ? 1 : 0, total);
    }

    // ---- head ----
    k_head<<<gNode, 256, 0, stream>>>(bufC, fcW, fcb, out, N);
}
